// DenseNetBackbone_42245298324156
// MI455X (gfx1250) — matrix operand order repaired and checked by reading, not compiled, not run on hardware
//
#include <hip/hip_runtime.h>
#include <hip/hip_bf16.h>
#include <math.h>

// ---------------------------------------------------------------------------
// CDNA5 (gfx1250) DenseNet backbone + deformable head.
// All convolutions run through one templated implicit-GEMM kernel:
//   - block = 8 wave32s, owns a 32-pixel tile (two 16-pixel B sub-tiles)
//   - 256 threads cooperatively build both 32x16 f16 im2col tiles in LDS
//   - each wave reads B via ds_load_b128, loads its weight rows with
//     unconditional global_load_b128 (clamped row, zero-selected after), and
//     issues two v_wmma_f32_16x16x32_f16 per K-step (A operand reused)
//   - BN(g*EPS)+b and ReLU fused into the im2col gather; bias in the epilogue
//   - kernel geometry (KH,KW,stride,pad) is compile-time -> no runtime division
// ---------------------------------------------------------------------------

typedef __attribute__((ext_vector_type(16))) _Float16 v16h;
typedef __attribute__((ext_vector_type(8)))  float    v8f;
typedef __attribute__((ext_vector_type(4)))  float    v4f;

#define EPS_SCALE 0.99999499987499375f   // 1/sqrt(1+1e-5)

template<int KH, int KW, int ST, int PD, bool ALIGNED>
__global__ __launch_bounds__(256)
void conv_wmma_t(const float* __restrict__ in, int CtotIn, int Cin, int H, int W,
                 const float* __restrict__ wgt, int Cout,
                 const float* __restrict__ bias,
                 const float* __restrict__ bng, const float* __restrict__ bnb, int relu,
                 float* __restrict__ out, int CtotOut, int coOff, int OH, int OW, int N,
                 int grpC)
{
    constexpr int KK = KH * KW;
    __shared__ alignas(32) _Float16 Bs[2][32 * 16];   // [sub-tile][lanehat][j]

    const int tid  = threadIdx.x;
    const int lane = tid & 31;
    const int wv   = tid >> 5;

    const int P      = N * OH * OW;
    const int tilesC = (Cout + 15) >> 4;
    const int grp    = blockIdx.x % grpC;
    const int pTile  = blockIdx.x / grpC;
    const int pBase  = pTile * 32;
    const int Ktot   = Cin * KK;

    // ---- producer mapping: thread fills Bs[s][lh*16 + jp .. +1]
    const int jp  = (tid * 2) & 15;      // even j
    const int lh  = tid >> 3;            // lanehat 0..31
    const int np  = lh & 15;             // B column (pixel slot)
    const int khp = lh >> 4;             // K half
    int pim[2] = {0, 0}, pyy[2] = {0, 0}, pxx[2] = {0, 0};
    bool pv[2] = {false, false};
#pragma unroll
    for (int s = 0; s < 2; ++s) {
        int pp = pBase + s * 16 + np;
        if (pp < P) { pv[s] = true; pim[s] = pp / (OH * OW);
                      int r = pp % (OH * OW); pyy[s] = r / OW; pxx[s] = r % OW; }
    }

    // ---- consumer mapping (per lane of each wave): two output pixels
    const int mn = lane & 15, hlc = lane >> 4;
    int cim[2] = {0, 0}, cyy[2] = {0, 0}, cxx[2] = {0, 0};
    bool cv[2] = {false, false};
#pragma unroll
    for (int s = 0; s < 2; ++s) {
        int pp = pBase + s * 16 + mn;
        if (pp < P) { cv[s] = true; cim[s] = pp / (OH * OW);
                      int r = pp % (OH * OW); cyy[s] = r / OW; cxx[s] = r % OW; }
    }

    const int  coTile = grp * 8 + wv;
    const bool doMM   = coTile < tilesC;
    const int  coBase = coTile * 16;
    const bool aok    = doMM && (coBase + mn) < Cout;
    const float* wrow = wgt + (size_t)(aok ? (coBase + mn) : 0) * Ktot;  // clamped: always valid

    v8f acc0 = {}, acc1 = {};
    for (int k0 = 0; k0 < Ktot; k0 += 32) {
        // -------- cooperative im2col: 2 K-elements x 2 sub-tiles per thread ----
#pragma unroll
        for (int u = 0; u < 2; ++u) {
            int j  = jp + u;
            int kk = k0 + j + khp * 8 + (j & 8);
            float v0 = 0.f, v1 = 0.f;
            if (ALIGNED || kk < Ktot) {
                int ci = kk / KK; int r = kk - ci * KK;      // compile-time KK
                int kh = r / KW;  int kw = r - kh * KW;
                float sc = 1.f, sb = 0.f;
                if (bng) { sc = bng[ci] * EPS_SCALE; sb = bnb[ci]; }
                const float* cbase = in + ((size_t)0 + ci) * H * W;
#pragma unroll
                for (int s = 0; s < 2; ++s) {
                    float v = 0.f;
                    if (pv[s]) {
                        int iy = pyy[s] * ST - PD + kh;
                        int ix = pxx[s] * ST - PD + kw;
                        if (iy >= 0 && iy < H && ix >= 0 && ix < W) {
                            v = cbase[((size_t)pim[s] * CtotIn) * H * W + (size_t)iy * W + ix];
                            if (bng)  v = v * sc + sb;
                            if (relu) v = v > 0.f ? v : 0.f;
                        }
                    }
                    if (s == 0) v0 = v; else v1 = v;
                }
            }
            Bs[0][lh * 16 + j] = (_Float16)v0;
            Bs[1][lh * 16 + j] = (_Float16)v1;
        }
        __syncthreads();

        if (doMM) {
            if (k0 + 32 < Ktot) __builtin_prefetch(wrow + k0 + 32, 0, 3); // global_prefetch_b8
            // ---- A operand: weight row, unconditional vector loads -----------
            v16h areg;
            if constexpr (ALIGNED) {
                const float* wp = wrow + k0 + hlc * 8;        // 16B-aligned (Ktot%32==0)
                v4f w0 = *(const v4f*)(wp);
                v4f w1 = *(const v4f*)(wp + 4);
                v4f w2 = *(const v4f*)(wp + 16);
                v4f w3 = *(const v4f*)(wp + 20);
#pragma unroll
                for (int e = 0; e < 4; ++e) {
                    areg[e]      = (_Float16)(aok ? w0[e] : 0.f);
                    areg[e + 4]  = (_Float16)(aok ? w1[e] : 0.f);
                    areg[e + 8]  = (_Float16)(aok ? w2[e] : 0.f);
                    areg[e + 12] = (_Float16)(aok ? w3[e] : 0.f);
                }
            } else {
#pragma unroll
                for (int j = 0; j < 16; ++j) {
                    int kk = k0 + ((j < 8) ? (j + hlc * 8) : (j + 8 + hlc * 8));
                    float v = (aok && kk < Ktot) ? wrow[kk] : 0.f;
                    areg[j] = (_Float16)v;
                }
            }
            // ---- B operands from LDS (ds_load_b128 x2 each) ------------------
            v16h b0 = *(const v16h*)&Bs[0][lane * 16];
            v16h b1 = *(const v16h*)&Bs[1][lane * 16];
            acc0 = __builtin_amdgcn_wmma_f32_16x16x32_f16(false, areg, false, b0,
                                                          (short)0, acc0, false, false);
            acc1 = __builtin_amdgcn_wmma_f32_16x16x32_f16(false, areg, false, b1,
                                                          (short)0, acc1, false, false);
        }
        __syncthreads();
    }

    // ---- epilogue: C VGPR r = row r (lanes 0..15) / r+8 (lanes 16..31)
    if (doMM) {
#pragma unroll
        for (int s = 0; s < 2; ++s) {
            if (!cv[s]) continue;
#pragma unroll
            for (int r = 0; r < 8; ++r) {
                int co = coBase + r + 8 * hlc;
                if (co < Cout) {
                    float v = (s == 0) ? acc0[r] : acc1[r];
                    if (bias) v += bias[co];
                    out[(((size_t)cim[s] * CtotOut + (co + coOff)) * OH + cyy[s]) * OW + cxx[s]] = v;
                }
            }
        }
    }
}

// ================= fused BN+ReLU + 3x3/s2 pad1 max-pool ====================
__global__ void bnrelu_maxpool_kernel(const float* __restrict__ in,
                                      const float* __restrict__ g, const float* __restrict__ b,
                                      float* __restrict__ out,
                                      int N, int C, int H, int W, int OH, int OW, int CtotOut)
{
    size_t gid = (size_t)blockIdx.x * blockDim.x + threadIdx.x;
    size_t total = (size_t)N * C * OH * OW;
    if (gid >= total) return;
    int ox = gid % OW; size_t t = gid / OW;
    int oy = t % OH;   t /= OH;
    int c  = t % C;    int n = t / C;
    float sc = g[c] * EPS_SCALE, sb = b[c];
    float m = -3.4e38f;
    const float* p = in + ((size_t)n * C + c) * H * W;
    for (int dy = 0; dy < 3; ++dy) {
        int iy = oy * 2 - 1 + dy;
        if (iy < 0 || iy >= H) continue;
        for (int dx = 0; dx < 3; ++dx) {
            int ix = ox * 2 - 1 + dx;
            if (ix < 0 || ix >= W) continue;
            float v = p[iy * W + ix] * sc + sb;
            v = v > 0.f ? v : 0.f;
            m = v > m ? v : m;
        }
    }
    out[(((size_t)n * CtotOut + c) * OH + oy) * OW + ox] = m;
}

// ============================ 2x2 avg-pool =================================
__global__ void avgpool2_kernel(const float* __restrict__ in, float* __restrict__ out,
                                int N, int C, int H, int W, int CtotIn, int CtotOut)
{
    int OH = H / 2, OW = W / 2;
    size_t gid = (size_t)blockIdx.x * blockDim.x + threadIdx.x;
    size_t total = (size_t)N * C * OH * OW;
    if (gid >= total) return;
    int ox = gid % OW; size_t t = gid / OW;
    int oy = t % OH;   t /= OH;
    int c  = t % C;    int n = t / C;
    const float* p = in + (((size_t)n * CtotIn + c) * H + oy * 2) * W + ox * 2;
    float v = 0.25f * (p[0] + p[1] + p[W] + p[W + 1]);
    out[(((size_t)n * CtotOut + c) * OH + oy) * OW + ox] = v;
}

// ============ deformable-conv sampling: bilinear gather + mask =============
__global__ void deform_prep_kernel(const float* __restrict__ x,
                                   const float* __restrict__ off,
                                   const float* __restrict__ msk,
                                   float* __restrict__ xo,
                                   int N, int C, int H, int W)
{
    size_t gid = (size_t)blockIdx.x * blockDim.x + threadIdx.x;
    size_t total = (size_t)N * C * H * W * 9;
    if (gid >= total) return;
    int i = gid % 9;  size_t t = gid / 9;
    int xw = t % W;   t /= W;
    int y  = t % H;   t /= H;
    int c  = t % C;   int n = t / C;
    int ky = i / 3, kx = i % 3;

    const int Hp = H + 2, Wp = W + 2;
    size_t hw = (size_t)H * W;
    float py = off[((size_t)n * 18 + i)     * hw + (size_t)y * W + xw] + (float)(y + 1) + (float)(ky - 1);
    float px = off[((size_t)n * 18 + 9 + i) * hw + (size_t)y * W + xw] + (float)(xw + 1) + (float)(kx - 1);
    float mv = msk[((size_t)n * 9 + i) * hw + (size_t)y * W + xw];
    mv = 1.f / (1.f + __expf(-mv));

    float fy = floorf(py), fx = floorf(px);
    float qy0 = fminf(fmaxf(fy,       0.f), (float)(Hp - 1));
    float qx0 = fminf(fmaxf(fx,       0.f), (float)(Wp - 1));
    float qy1 = fminf(fmaxf(fy + 1.f, 0.f), (float)(Hp - 1));
    float qx1 = fminf(fmaxf(fx + 1.f, 0.f), (float)(Wp - 1));
    py = fminf(fmaxf(py, 0.f), (float)(Hp - 1));
    px = fminf(fmaxf(px, 0.f), (float)(Wp - 1));
    float g_lt = (1.f + (qy0 - py)) * (1.f + (qx0 - px));
    float g_rb = (1.f - (qy1 - py)) * (1.f - (qx1 - px));
    float g_lb = (1.f + (qy0 - py)) * (1.f - (qx1 - px));
    float g_rt = (1.f - (qy1 - py)) * (1.f + (qx0 - px));

    const float* xc = x + ((size_t)n * C + c) * hw;
    auto samp = [&](float qy, float qx) -> float {
        int iy = (int)qy - 1, ix = (int)qx - 1;   // padded -> original coords
        if (iy < 0 || iy >= H || ix < 0 || ix >= W) return 0.f;
        return xc[(size_t)iy * W + ix];
    };
    float v = g_lt * samp(qy0, qx0) + g_rb * samp(qy1, qx1)
            + g_lb * samp(qy0, qx1) + g_rt * samp(qy1, qx0);
    v *= mv;
    xo[(((size_t)n * C + c) * (3 * H) + (size_t)(y * 3 + ky)) * (3 * W) + (xw * 3 + kx)] = v;
}

// ============== transposed conv: k=3, stride=2, pad=1 ======================
// w shape (Cin, Cout, 3, 3); wf[co][ci][kh][kw] = w[ci][co][2-kh][2-kw]
__global__ void deconv_kernel(const float* __restrict__ x, const float* __restrict__ w,
                              const float* __restrict__ bias, float* __restrict__ out,
                              int N, int Cin, int Cout, int H, int W, int OH, int OW)
{
    size_t gid = (size_t)blockIdx.x * blockDim.x + threadIdx.x;
    size_t total = (size_t)N * Cout * OH * OW;
    if (gid >= total) return;
    int ox = gid % OW; size_t t = gid / OW;
    int oy = t % OH;   t /= OH;
    int co = t % Cout; int n = t / Cout;

    float acc = bias[co];
    for (int kh = 0; kh < 3; ++kh) {
        int dy = oy + kh - 1;
        if (dy < 0 || (dy & 1) || (dy >> 1) >= H) continue;
        int iy = dy >> 1;
        for (int kw = 0; kw < 3; ++kw) {
            int dx = ox + kw - 1;
            if (dx < 0 || (dx & 1) || (dx >> 1) >= W) continue;
            int ix = dx >> 1;
            for (int ci = 0; ci < Cin; ++ci) {
                float xv = x[(((size_t)n * Cin + ci) * H + iy) * W + ix];
                float wv = w[(((size_t)ci * Cout + co) * 3 + (2 - kh)) * 3 + (2 - kw)];
                acc = fmaf(xv, wv, acc);
            }
        }
    }
    out[(((size_t)n * Cout + co) * OH + oy) * OW + ox] = acc;
}

// ================= bilinear resize, align_corners=True =====================
__global__ void resize_bilinear_kernel(const float* __restrict__ in, float* __restrict__ out,
                                       int N, int C, int H, int W, int OH, int OW)
{
    size_t gid = (size_t)blockIdx.x * blockDim.x + threadIdx.x;
    size_t total = (size_t)N * C * OH * OW;
    if (gid >= total) return;
    int ox = gid % OW; size_t t = gid / OW;
    int oy = t % OH;   t /= OH;
    int c  = t % C;    int n = t / C;
    float sy = (OH > 1) ? oy * ((float)(H - 1) / (float)(OH - 1)) : 0.f;
    float sx = (OW > 1) ? ox * ((float)(W - 1) / (float)(OW - 1)) : 0.f;
    int y0 = (int)floorf(sy); y0 = y0 < 0 ? 0 : (y0 > H - 2 ? H - 2 : y0);
    int x0 = (int)floorf(sx); x0 = x0 < 0 ? 0 : (x0 > W - 2 ? W - 2 : x0);
    float ty = sy - (float)y0, tx = sx - (float)x0;
    const float* p = in + ((size_t)n * C + c) * H * W;
    float v = (1.f - ty) * ((1.f - tx) * p[(size_t)y0 * W + x0] + tx * p[(size_t)y0 * W + x0 + 1])
            + ty         * ((1.f - tx) * p[(size_t)(y0 + 1) * W + x0] + tx * p[(size_t)(y0 + 1) * W + x0 + 1]);
    out[(((size_t)n * C + c) * OH + oy) * OW + ox] = v;
}

// ===========================================================================
extern "C" void kernel_launch(void* const* d_in, const int* in_sizes, int n_in,
                              void* d_out, int out_size, void* d_ws, size_t ws_size,
                              hipStream_t stream)
{
    (void)in_sizes; (void)n_in; (void)out_size; (void)ws_size;
    const int N = 16;

    // -------- input index map (setup_inputs insertion order, recursively) ----
    // 0:x 1:conv0_w 2:norm0_g 3:norm0_b
    // block1 layer i (i=0..5) at base 4+6i: n1g,n1b,c1w,n2g,n2b,c2w
    // 40:trans_n_g 41:trans_n_b 42:trans_c_w
    // block2 layer i (i=0..11) at base 43+6i: n1g,n1b,c1w,n2g,n2b,c2w
    // 115:layer1_w 116:layer1_b
    // deform1 117..121: p_w,p_b,m_w,m_b,w ; deform2 122..126 ; deform3 127..131
    // 132:dc1_w 133:dc1_b 134:dc2_w 135:dc2_b 136:dc3_w 137:dc3_b
    auto F = [&](int i) { return (const float*)d_in[i]; };

    // -------- workspace bump allocator --------------------------------------
    char* ws = (char*)d_ws;
    size_t cur = 0;
    auto alloc = [&](size_t elems) -> float* {
        float* p = (float*)(ws + cur);
        cur += ((elems * 4 + 255) / 256) * 256;
        return p;
    };
    float* conv0o = alloc((size_t)N * 64 * 112 * 112);
    float* blk1   = alloc((size_t)N * 256 * 56 * 56);
    float* t1     = alloc((size_t)N * 128 * 56 * 56);
    float* blk2   = alloc((size_t)N * 512 * 28 * 28);
    float* t2     = alloc((size_t)N * 128 * 28 * 28);
    float* l1     = alloc((size_t)N * 256 * 28 * 28);
    float* offb   = alloc((size_t)N * 18 * 28 * 28);
    float* mskb   = alloc((size_t)N * 9 * 28 * 28);
    float* xob    = alloc((size_t)N * 256 * 84 * 84);
    float* d1b    = alloc((size_t)N * 128 * 28 * 28);
    float* d2b    = alloc((size_t)N * 128 * 28 * 28);
    float* d3b    = alloc((size_t)N * 64 * 28 * 28);
    float* up1    = alloc((size_t)N * 64 * 55 * 55);
    float* up2    = alloc((size_t)N * 128 * 109 * 109);
    float* rs1    = alloc((size_t)N * 128 * 112 * 112);
    float* up3    = alloc((size_t)N * 1 * 223 * 223);

    // kind: 0 = 1x1/s1/p0, 1 = 3x3/s1/p1, 2 = 3x3/s3/p0, 3 = 7x7/s2/p3
    auto conv = [&](int kind, const float* in, int CtotIn, int Cin, int H, int W,
                    const float* wgt, int Cout, const float* bias,
                    const float* g, const float* b, int relu,
                    float* out, int CtotOut, int coOff, int OH, int OW) {
        int P = N * OH * OW;
        int tilesP = (P + 31) / 32;               // 32 pixels per block
        int tilesC = (Cout + 15) / 16;
        int grpC   = (tilesC + 7) / 8;
        int blocks = tilesP * grpC;
        switch (kind) {
        case 0: conv_wmma_t<1, 1, 1, 0, true ><<<blocks, 256, 0, stream>>>(
                    in, CtotIn, Cin, H, W, wgt, Cout, bias, g, b, relu,
                    out, CtotOut, coOff, OH, OW, N, grpC); break;
        case 1: conv_wmma_t<3, 3, 1, 1, true ><<<blocks, 256, 0, stream>>>(
                    in, CtotIn, Cin, H, W, wgt, Cout, bias, g, b, relu,
                    out, CtotOut, coOff, OH, OW, N, grpC); break;
        case 2: conv_wmma_t<3, 3, 3, 0, true ><<<blocks, 256, 0, stream>>>(
                    in, CtotIn, Cin, H, W, wgt, Cout, bias, g, b, relu,
                    out, CtotOut, coOff, OH, OW, N, grpC); break;
        default: conv_wmma_t<7, 7, 2, 3, false><<<blocks, 256, 0, stream>>>(
                    in, CtotIn, Cin, H, W, wgt, Cout, bias, g, b, relu,
                    out, CtotOut, coOff, OH, OW, N, grpC); break;
        }
    };
    auto ew = [&](size_t total) { return dim3((unsigned)((total + 255) / 256)); };

    // ---- stem: conv0 7x7/s2 pad3 -> fused BN+ReLU+maxpool into blk1[ch 0:64)
    conv(3, F(0), 3, 3, 224, 224, F(1), 64, nullptr, nullptr, nullptr, 0,
         conv0o, 64, 0, 112, 112);
    bnrelu_maxpool_kernel<<<ew((size_t)N * 64 * 56 * 56), 256, 0, stream>>>(
        conv0o, F(2), F(3), blk1, N, 64, 112, 112, 56, 56, 256);

    // ---- dense block 1 (6 layers, growth 32, concat via channel offsets)
    for (int i = 0; i < 6; ++i) {
        int base = 4 + 6 * i, cin = 64 + 32 * i;
        conv(0, blk1, 256, cin, 56, 56, F(base + 2), 128, nullptr, F(base + 0), F(base + 1), 1,
             t1, 128, 0, 56, 56);
        conv(1, t1, 128, 128, 56, 56, F(base + 5), 32, nullptr, F(base + 3), F(base + 4), 1,
             blk1, 256, 64 + 32 * i, 56, 56);
    }

    // ---- transition 1: BN+ReLU + 1x1 conv -> avgpool into blk2[ch 0:128)
    conv(0, blk1, 256, 256, 56, 56, F(42), 128, nullptr, F(40), F(41), 1,
         t1, 128, 0, 56, 56);
    avgpool2_kernel<<<ew((size_t)N * 128 * 28 * 28), 256, 0, stream>>>(
        t1, blk2, N, 128, 56, 56, 128, 512);

    // ---- dense block 2 (12 layers)
    for (int i = 0; i < 12; ++i) {
        int base = 43 + 6 * i, cin = 128 + 32 * i;
        conv(0, blk2, 512, cin, 28, 28, F(base + 2), 128, nullptr, F(base + 0), F(base + 1), 1,
             t2, 128, 0, 28, 28);
        conv(1, t2, 128, 128, 28, 28, F(base + 5), 32, nullptr, F(base + 3), F(base + 4), 1,
             blk2, 512, 128 + 32 * i, 28, 28);
    }

    // ---- layer1: 3x3 conv 512->256, bias, no activation
    conv(1, blk2, 512, 512, 28, 28, F(115), 256, F(116), nullptr, nullptr, 0,
         l1, 256, 0, 28, 28);

    // ---- deformable convs: offset/mask convs (WMMA) + gather + stride-3 WMMA conv
    auto deform = [&](const float* x, int Cin, int Cout, int pbase, float* out) {
        conv(1, x, Cin, Cin, 28, 28, F(pbase + 0), 18, F(pbase + 1), nullptr, nullptr, 0,
             offb, 18, 0, 28, 28);
        conv(1, x, Cin, Cin, 28, 28, F(pbase + 2), 9, F(pbase + 3), nullptr, nullptr, 0,
             mskb, 9, 0, 28, 28);
        deform_prep_kernel<<<ew((size_t)N * Cin * 28 * 28 * 9), 256, 0, stream>>>(
            x, offb, mskb, xob, N, Cin, 28, 28);
        conv(2, xob, Cin, Cin, 84, 84, F(pbase + 4), Cout, nullptr, nullptr, nullptr, 0,
             out, Cout, 0, 28, 28);
    };
    deform(l1,  256, 128, 117, d1b);
    deform(d1b, 128, 128, 122, d2b);
    deform(d2b, 128,  64, 127, d3b);

    // ---- decoder: 3x transposed conv (k3 s2 p1) + align-corners resizes
    deconv_kernel<<<ew((size_t)N * 64 * 55 * 55), 256, 0, stream>>>(
        d3b, F(132), F(133), up1, N, 64, 64, 28, 28, 55, 55);
    deconv_kernel<<<ew((size_t)N * 128 * 109 * 109), 256, 0, stream>>>(
        up1, F(134), F(135), up2, N, 64, 128, 55, 55, 109, 109);
    resize_bilinear_kernel<<<ew((size_t)N * 128 * 112 * 112), 256, 0, stream>>>(
        up2, rs1, N, 128, 109, 109, 112, 112);
    deconv_kernel<<<ew((size_t)N * 1 * 223 * 223), 256, 0, stream>>>(
        rs1, F(136), F(137), up3, N, 128, 1, 112, 112, 223, 223);
    resize_bilinear_kernel<<<ew((size_t)N * 1 * 224 * 224), 256, 0, stream>>>(
        up3, (float*)d_out, N, 1, 223, 223, 224, 224);
}
